// IRBlock_2456721293548
// MI455X (gfx1250) — matrix operand order repaired and checked by reading, not compiled, not run on hardware
//
#include <hip/hip_runtime.h>

#define DEV __device__ __forceinline__

typedef _Float16 half_t;
typedef half_t v16h __attribute__((ext_vector_type(16)));
typedef half_t v8h  __attribute__((ext_vector_type(8)));
typedef half_t v4h  __attribute__((ext_vector_type(4)));
typedef float  v8f  __attribute__((ext_vector_type(8)));

constexpr int B_   = 2;
constexpr int DIM  = 192;
constexpr int NH   = 6;
constexpr int HD   = 32;     // head dim == WMMA K
constexpr int WIN  = 8;
constexpr int WS   = 64;     // tokens per window
constexpr int RH   = 32;
constexpr int RW   = 32;
constexpr int L    = RH * RW;      // 1024 windows
constexpr int HPIX = 256, WPIX = 256;
constexpr float SCALE = 0.17677669529663687f;  // 32^-0.5

// ---------------- WMMA fragment loaders (wave32, gfx1250 layouts) ----------
// A 16x32 f16: lane m = lane&15; VGPR0-3 hold K = (lane<16?0:8)+0..7,
// VGPR4-7 hold K = 16 + (lane<16?0:8)+0..7.
DEV v16h load_a16(const half_t* base, int ld, int m0, int k0, int lane) {
  int m  = m0 + (lane & 15);
  int kb = (lane < 16) ? 0 : 8;
  const half_t* p = base + m * ld + k0;
  v8h lo = *(const v8h*)(p + kb);
  v8h hi = *(const v8h*)(p + 16 + kb);
  return __builtin_shufflevector(lo, hi, 0,1,2,3,4,5,6,7,8,9,10,11,12,13,14,15);
}
// B 32x16 f16 with B[k][n] = M[n0+n][k0+k] (M row-major, contiguous K):
// lane n = lane&15; elems 0-7 hold K = (lane<16?0:8)+0..7, elems 8-15 hold K = 16 + (lane<16?0:8)+0..7.
DEV v16h load_b16(const half_t* base, int ld, int n0, int k0, int lane) {
  int n  = n0 + (lane & 15);
  int kb = (lane < 16) ? 0 : 8;
  const half_t* p = base + n * ld + k0;
  v8h lo = *(const v8h*)(p + kb);
  v8h hi = *(const v8h*)(p + 16 + kb);
  return __builtin_shufflevector(lo, hi, 0,1,2,3,4,5,6,7,8,9,10,11,12,13,14,15);
}

// ---------------- Kernel 1: 8x8 mean-pool q,k -> qr,kr [b][l][c] -----------
__global__ void pool_qk(const float* __restrict__ x,
                        float* __restrict__ qr, float* __restrict__ kr) {
  int o = blockIdx.x * blockDim.x + threadIdx.x;
  if (o >= B_ * 2 * DIM * L) return;
  int l = o & (L - 1);
  int c = (o >> 10) % (2 * DIM);
  int b = o / (2 * DIM * L);
  int rh = l >> 5, rw = l & 31;
  const float* p = x + (((size_t)(b * 3 * DIM + c) * HPIX) + rh * WIN) * WPIX + rw * WIN;
  float s = 0.f;
  #pragma unroll
  for (int r = 0; r < WIN; ++r) {
    const float4* p4 = (const float4*)(p + r * WPIX);
    float4 a = p4[0], bb = p4[1];
    s += a.x + a.y + a.z + a.w + bb.x + bb.y + bb.z + bb.w;
  }
  s *= (1.f / 64.f);
  if (c < DIM) qr[((size_t)b * L + l) * DIM + c]         = s;
  else         kr[((size_t)b * L + l) * DIM + (c - DIM)] = s;
}

// ---------------- Kernel 2: router argmax over 9 reflected neighbors -------
DEV int reflect1(int p, int n) { return p < 0 ? -p : (p >= n ? 2 * n - 2 - p : p); }

__global__ void router(const float* __restrict__ qr, const float* __restrict__ kr,
                       int* __restrict__ idx_dir, int* __restrict__ idx_ref) {
  int t = blockIdx.x * blockDim.x + threadIdx.x;
  if (t >= B_ * L) return;
  int b = t / L, l = t & (L - 1);
  int rh = l >> 5, rw = l & 31;
  const float* qp = qr + ((size_t)b * L + l) * DIM;
  float best = -1e30f; int bn = 0, bl = 0;
  for (int n = 0; n < 9; ++n) {
    int r2 = reflect1(rh + n / 3 - 1, RH);
    int c2 = reflect1(rw + n % 3 - 1, RW);
    int nl = r2 * RW + c2;
    const float* kp = kr + ((size_t)b * L + nl) * DIM;
    float dot = 0.f;
    for (int c = 0; c < DIM; c += 4) {
      float4 a  = *(const float4*)(qp + c);
      float4 kk = *(const float4*)(kp + c);
      dot += a.x * kk.x + a.y * kk.y + a.z * kk.z + a.w * kk.w;
    }
    if (n == 4) dot -= 100.f;
    if (dot > best) { best = dot; bn = n; bl = nl; }   // first-max wins
  }
  idx_dir[t] = bn;
  idx_ref[t] = bl;
}

// ---------------- Kernel 3: WMMA attention, one block per (b,l,h) ----------
__launch_bounds__(256, 1)
__global__ void attn_kernel(const float* __restrict__ x, const float* __restrict__ pos,
                            const int* __restrict__ idx_dir, const int* __restrict__ idx_ref,
                            float* __restrict__ out) {
  __shared__ __align__(32) half_t Qs[WS * HD];          // 64x32 [s][d]
  __shared__ __align__(32) half_t Kg[2 * WS * HD];      // 128x32 [t][d]
  __shared__ __align__(32) half_t Vt[HD * 2 * WS];      // 32x128 [d][t] (transposed)
  __shared__ __align__(32) float  Sm[WS * 2 * WS];      // 64x128 scores
  __shared__ __align__(32) half_t Pm[WS * 2 * WS];      // 64x128 exp probs (unnormalized)
  __shared__ __align__(32) float  Om[WS * HD];          // 64x32
  __shared__ float PosSelf[225];
  __shared__ float PosDir[225];
  __shared__ float red[256];
  __shared__ float rowmax[WS];
  __shared__ float rowrcp[WS];

  const int tid  = threadIdx.x;
  const int lane = tid & 31;
  const int w    = tid >> 5;

  int bx = blockIdx.x;
  int h  = bx % NH;
  int l  = (bx / NH) % L;
  int b  = bx / (NH * L);
  int rh  = l >> 5, rw = l & 31;
  int lref = idx_ref[b * L + l];
  int rh2  = lref >> 5, rw2 = lref & 31;
  int ndir = idx_dir[b * L + l];

  // ---- stage the two relative-pos tables needed by this block ------------
  if (tid < 225) {
    PosSelf[tid] = pos[(4    * NH + h) * 225 + tid];
    PosDir[tid]  = pos[(ndir * NH + h) * 225 + tid];
  }

  // ---- stage tiles into LDS as f16, float4 (4 window cols) per load ------
  #pragma unroll
  for (int it = 0; it < 2; ++it) {
    int g   = tid + it * 256;          // 0..511
    int d   = g >> 4;                  // 0..31
    int rem = g & 15;
    int pr  = rem >> 1;                // window row 0..7
    int hc  = rem & 1;                 // col half 0..1
    int s0  = pr * WIN + hc * 4;       // first of 4 consecutive tokens
    size_t a;
    float4 v;
    // Q self -> Qs[s][d]
    a = ((size_t)(b * 3 * DIM + 0 * DIM + h * HD + d) * HPIX + rh * WIN + pr) * WPIX + rw * WIN + hc * 4;
    v = *(const float4*)&x[a];
    Qs[(s0 + 0) * HD + d] = (half_t)v.x; Qs[(s0 + 1) * HD + d] = (half_t)v.y;
    Qs[(s0 + 2) * HD + d] = (half_t)v.z; Qs[(s0 + 3) * HD + d] = (half_t)v.w;
    // K self -> Kg[t][d]
    a = ((size_t)(b * 3 * DIM + 1 * DIM + h * HD + d) * HPIX + rh * WIN + pr) * WPIX + rw * WIN + hc * 4;
    v = *(const float4*)&x[a];
    Kg[(s0 + 0) * HD + d] = (half_t)v.x; Kg[(s0 + 1) * HD + d] = (half_t)v.y;
    Kg[(s0 + 2) * HD + d] = (half_t)v.z; Kg[(s0 + 3) * HD + d] = (half_t)v.w;
    // K dir -> Kg[64+t][d]
    a = ((size_t)(b * 3 * DIM + 1 * DIM + h * HD + d) * HPIX + rh2 * WIN + pr) * WPIX + rw2 * WIN + hc * 4;
    v = *(const float4*)&x[a];
    Kg[(WS + s0 + 0) * HD + d] = (half_t)v.x; Kg[(WS + s0 + 1) * HD + d] = (half_t)v.y;
    Kg[(WS + s0 + 2) * HD + d] = (half_t)v.z; Kg[(WS + s0 + 3) * HD + d] = (half_t)v.w;
    // V self -> Vt[d][t] (contiguous 4 halves -> one 8B DS store)
    a = ((size_t)(b * 3 * DIM + 2 * DIM + h * HD + d) * HPIX + rh * WIN + pr) * WPIX + rw * WIN + hc * 4;
    v = *(const float4*)&x[a];
    { v4h p4 = { (half_t)v.x, (half_t)v.y, (half_t)v.z, (half_t)v.w };
      *(v4h*)&Vt[d * (2 * WS) + s0] = p4; }
    // V dir -> Vt[d][64+t]
    a = ((size_t)(b * 3 * DIM + 2 * DIM + h * HD + d) * HPIX + rh2 * WIN + pr) * WPIX + rw2 * WIN + hc * 4;
    v = *(const float4*)&x[a];
    { v4h p4 = { (half_t)v.x, (half_t)v.y, (half_t)v.z, (half_t)v.w };
      *(v4h*)&Vt[d * (2 * WS) + WS + s0] = p4; }
  }
  __syncthreads();

  // ---- S = Q @ Kg^T : wave w -> M-tile (w&3), N-half (w>>2), 4 WMMAs -----
  {
    int mt  = (w & 3) << 4;
    int nh2 = (w >> 2) << 6;
    v16h a = load_a16(Qs, HD, mt, 0, lane);
    int cl = lane & 15, ro = (lane < 16) ? 0 : 8;
    #pragma unroll
    for (int j = 0; j < 4; ++j) {
      int nt = nh2 + j * 16;
      v16h bf = load_b16(Kg, HD, nt, 0, lane);
      v8f c = {};
      c = __builtin_amdgcn_wmma_f32_16x16x32_f16(false, a, false, bf, (short)0, c, false, false);
      #pragma unroll
      for (int r = 0; r < 8; ++r)
        Sm[(mt + r + ro) * (2 * WS) + nt + cl] = c[r];
    }
  }
  __syncthreads();

  // ---- scale + relative-pos bias + softmax (row = tid/4, 32 cols each) ---
  {
    int r  = tid >> 2;
    int c0 = (tid & 3) << 5;
    int sr = r >> 3, sc = r & 7;
    const float* ptab = (c0 < WS) ? PosSelf : PosDir;   // uniform per thread
    float lmax = -1e30f;
    for (int c = 0; c < 32; ++c) {
      int t  = c0 + c;
      int tt = t & 63;
      int pidx = (sr - (tt >> 3) + 7) * 15 + (sc - (tt & 7) + 7);
      float av = Sm[r * (2 * WS) + t] * SCALE + ptab[pidx];
      Sm[r * (2 * WS) + t] = av;
      lmax = fmaxf(lmax, av);
    }
    red[tid] = lmax;
    __syncthreads();
    if (tid < WS)
      rowmax[tid] = fmaxf(fmaxf(red[tid*4], red[tid*4+1]), fmaxf(red[tid*4+2], red[tid*4+3]));
    __syncthreads();
    float rm = rowmax[r];
    float lsum = 0.f;
    for (int c = 0; c < 32; ++c) {
      int t = c0 + c;
      float ex = __expf(Sm[r * (2 * WS) + t] - rm);
      Pm[r * (2 * WS) + t] = (half_t)ex;
      lsum += ex;
    }
    red[tid] = lsum;
    __syncthreads();
    if (tid < WS)
      rowrcp[tid] = 1.f / (red[tid*4] + red[tid*4+1] + red[tid*4+2] + red[tid*4+3]);
  }
  __syncthreads();

  // ---- O = P @ Vg : wave w -> tile (M = w&3, N = w>>2), 4 chained WMMAs --
  {
    int mt = (w & 3) << 4;
    int d0 = (w >> 2) << 4;
    v8f c = {};
    #pragma unroll
    for (int kc = 0; kc < 4; ++kc) {
      int tk0 = kc << 5;
      v16h a  = load_a16(Pm, 2 * WS, mt, tk0, lane);
      v16h bf = load_b16(Vt, 2 * WS, d0, tk0, lane);
      c = __builtin_amdgcn_wmma_f32_16x16x32_f16(false, a, false, bf, (short)0, c, false, false);
    }
    int cl = lane & 15, ro = (lane < 16) ? 0 : 8;
    #pragma unroll
    for (int r = 0; r < 8; ++r)
      Om[(mt + r + ro) * HD + d0 + cl] = c[r];
  }
  __syncthreads();

  // ---- normalize by row sum, write out as float4 (NCHW layout) -----------
  #pragma unroll
  for (int it = 0; it < 2; ++it) {
    int g   = tid + it * 256;          // 0..511
    int d   = g >> 4;
    int rem = g & 15;
    int pr  = rem >> 1;
    int hc  = rem & 1;
    int s0  = pr * WIN + hc * 4;
    float4 v;
    v.x = Om[(s0 + 0) * HD + d] * rowrcp[s0 + 0];
    v.y = Om[(s0 + 1) * HD + d] * rowrcp[s0 + 1];
    v.z = Om[(s0 + 2) * HD + d] * rowrcp[s0 + 2];
    v.w = Om[(s0 + 3) * HD + d] * rowrcp[s0 + 3];
    size_t oi = ((size_t)(b * DIM + h * HD + d) * HPIX + rh * WIN + pr) * WPIX + rw * WIN + hc * 4;
    *(float4*)&out[oi] = v;
  }
}

// ---------------------------------------------------------------------------
extern "C" void kernel_launch(void* const* d_in, const int* in_sizes, int n_in,
                              void* d_out, int out_size, void* d_ws, size_t ws_size,
                              hipStream_t stream) {
  (void)in_sizes; (void)n_in; (void)out_size; (void)ws_size;
  const float* x   = (const float*)d_in[0];
  const float* pos = (const float*)d_in[1];
  float* out = (float*)d_out;

  float* qr = (float*)d_ws;                       // [B][L][192]
  float* kr = qr + (size_t)B_ * L * DIM;          // [B][L][192]
  int* idxd = (int*)(kr + (size_t)B_ * L * DIM);  // [B*L]
  int* idxr = idxd + B_ * L;                      // [B*L]

  int poolN = B_ * 2 * DIM * L;
  pool_qk<<<(poolN + 255) / 256, 256, 0, stream>>>(x, qr, kr);
  router<<<(B_ * L + 255) / 256, 256, 0, stream>>>(qr, kr, idxd, idxr);
  attn_kernel<<<B_ * L * NH, 256, 0, stream>>>(x, pos, idxd, idxr, out);
}
